// AttentionBlock_4131758538829
// MI455X (gfx1250) — compile-verified, matrix operand order checked by reading, not run on hardware
//
#include <hip/hip_runtime.h>
#include <hip/hip_bf16.h>

// Problem constants
#define B_  32
#define S_  512
#define D_  768
#define H_  12
#define HD_ 64
#define FF_ 3072
#define TD_ 2304   // 3*D

typedef __attribute__((ext_vector_type(16))) __bf16 v16bf;
typedef __attribute__((ext_vector_type(8)))  float  v8f;

struct Frag32B { uint4 lo, hi; };

// float -> bf16 round-to-nearest-even
static __device__ __forceinline__ unsigned short f2bf(float f) {
    unsigned int u = __builtin_bit_cast(unsigned int, f);
    unsigned int r = u + 0x7FFFu + ((u >> 16) & 1u);
    return (unsigned short)(r >> 16);
}

// Load a 16-element bf16 A/B fragment slice for this lane.
// p must already point at (base + 8*sub): lo = k[8s..8s+7], hi = k[16+8s..16+8s+7]
static __device__ __forceinline__ v16bf load_frag_g(const unsigned short* __restrict__ p) {
    Frag32B t;
    t.lo = *reinterpret_cast<const uint4*>(p);
    t.hi = *reinterpret_cast<const uint4*>(p + 16);
    return __builtin_bit_cast(v16bf, t);
}

// ---- DPP16 cross-lane reductions over each 16-lane half (VALU only, no LDS) ----
template<int CTRL>
static __device__ __forceinline__ float dpp_mv(float x) {
    return __builtin_bit_cast(float,
        __builtin_amdgcn_update_dpp(0, __builtin_bit_cast(int, x), CTRL, 0xF, 0xF, true));
}
static __device__ __forceinline__ float red16_max(float v) {
    v = fmaxf(v, dpp_mv<0xB1>(v));    // quad_perm [1,0,3,2]  (xor 1)
    v = fmaxf(v, dpp_mv<0x4E>(v));    // quad_perm [2,3,0,1]  (xor 2)
    v = fmaxf(v, dpp_mv<0x141>(v));   // row_half_mirror      (combine quads in 8-group)
    v = fmaxf(v, dpp_mv<0x140>(v));   // row_mirror           (combine 8-groups in 16)
    return v;
}
static __device__ __forceinline__ float red16_sum(float v) {
    v += dpp_mv<0xB1>(v);
    v += dpp_mv<0x4E>(v);
    v += dpp_mv<0x141>(v);
    v += dpp_mv<0x140>(v);
    return v;
}

// ---------------------------------------------------------------- cast f32->bf16
__global__ void cast_f32_bf16(const float* __restrict__ in,
                              unsigned short* __restrict__ out, int n) {
    int i = blockIdx.x * 256 + threadIdx.x;
    if (i < n) out[i] = f2bf(in[i]);
}

// ---------------------------------------------------------------- layernorm -> bf16
__global__ __launch_bounds__(256) void layernorm_bf16(
    const float* __restrict__ x, const float* __restrict__ g,
    const float* __restrict__ b, unsigned short* __restrict__ out)
{
    __shared__ float red[256];
    const int row = blockIdx.x;
    const int t = threadIdx.x;
    const float* xr = x + (size_t)row * D_;
    float v0 = xr[t], v1 = xr[t + 256], v2 = xr[t + 512];
    red[t] = v0 + v1 + v2;
    __syncthreads();
    for (int off = 128; off > 0; off >>= 1) {
        if (t < off) red[t] += red[t + off];
        __syncthreads();
    }
    const float mu = red[0] * (1.0f / D_);
    __syncthreads();
    float d0 = v0 - mu, d1 = v1 - mu, d2 = v2 - mu;
    red[t] = d0 * d0 + d1 * d1 + d2 * d2;
    __syncthreads();
    for (int off = 128; off > 0; off >>= 1) {
        if (t < off) red[t] += red[t + off];
        __syncthreads();
    }
    const float rstd = rsqrtf(red[0] * (1.0f / D_) + 1e-5f);
    unsigned short* o = out + (size_t)row * D_;
    o[t]       = f2bf(d0 * rstd * g[t]       + b[t]);
    o[t + 256] = f2bf(d1 * rstd * g[t + 256] + b[t + 256]);
    o[t + 512] = f2bf(d2 * rstd * g[t + 512] + b[t + 512]);
}

// ---------------------------------------------------------------- V transpose: [B,S,3D] -> [B,H,HD,S]
__global__ void transpose_v(const unsigned short* __restrict__ qkv,
                            unsigned short* __restrict__ vt) {
    int i = blockIdx.x * 256 + threadIdx.x;   // i < B*H*HD*S
    int s  = i & (S_ - 1);
    int t  = i >> 9;
    int d  = t & (HD_ - 1);
    int bh = t >> 6;
    int h  = bh % H_;
    int b  = bh / H_;
    vt[i] = qkv[(size_t)(b * S_ + s) * TD_ + 2 * D_ + h * HD_ + d];
}

// ---------------------------------------------------------------- bf16 WMMA GEMM: C = A[M,K] * W[N,K]^T
// Wave tile 64x64 (16 WMMAs / k-step), block = 4 waves -> 128x128 tile.
// EPI 0: +bias -> bf16 out ; EPI 1: +bias +resid -> f32 out ; EPI 2: +bias, GELU -> bf16 out
template<int EPI>
__global__ __launch_bounds__(128) void gemm_bf16_wmma(
    const unsigned short* __restrict__ A, const unsigned short* __restrict__ W,
    const float* __restrict__ bias, const float* __restrict__ resid,
    float* __restrict__ outF, unsigned short* __restrict__ outB,
    int M, int N, int K)
{
    const int tid = threadIdx.x;
    const int ln = tid & 15, sub = (tid >> 4) & 1, w = tid >> 5;
    const int mb = blockIdx.y * 128 + (w >> 1) * 64;
    const int nb = blockIdx.x * 128 + (w & 1) * 64;

    v8f acc[4][4] = {};
    for (int k0 = 0; k0 < K; k0 += 32) {
        if (k0 + 32 < K)
            __builtin_prefetch(A + (size_t)(mb + ln) * K + k0 + 32, 0, 1);
        v16bf af[4], bf[4];
#pragma unroll
        for (int t = 0; t < 4; ++t) {
            af[t] = load_frag_g(A + (size_t)(mb + t * 16 + ln) * K + k0 + 8 * sub);
            bf[t] = load_frag_g(W + (size_t)(nb + t * 16 + ln) * K + k0 + 8 * sub);
        }
#pragma unroll
        for (int mt = 0; mt < 4; ++mt)
#pragma unroll
            for (int nt = 0; nt < 4; ++nt)
                acc[mt][nt] = __builtin_amdgcn_wmma_f32_16x16x32_bf16(
                    false, af[mt], false, bf[nt], (short)0, acc[mt][nt], false, false);
    }
#pragma unroll
    for (int mt = 0; mt < 4; ++mt) {
#pragma unroll
        for (int nt = 0; nt < 4; ++nt) {
            const int n = nb + nt * 16 + ln;
            const float bv = bias[n];
#pragma unroll
            for (int r = 0; r < 8; ++r) {
                const int m = mb + mt * 16 + sub * 8 + r;
                float v = acc[mt][nt][r] + bv;
                const size_t off = (size_t)m * N + n;
                if constexpr (EPI == 0) {
                    outB[off] = f2bf(v);
                } else if constexpr (EPI == 1) {
                    outF[off] = v + resid[off];
                } else {
                    outB[off] = f2bf(0.5f * v * (1.0f + erff(v * 0.70710678118f)));
                }
            }
        }
    }
}

// ---------------------------------------------------------------- flash attention (one wave per 16-row q tile)
__global__ __launch_bounds__(128) void attn_wmma(
    const unsigned short* __restrict__ qkv,
    const unsigned short* __restrict__ vt,
    const float* __restrict__ ebias,
    unsigned short* __restrict__ o)
{
    __shared__ __align__(16) unsigned short sP[4][16 * 32];
    const int tid = threadIdx.x;
    const int ln = tid & 15, sub = (tid >> 4) & 1, w = tid >> 5;
    const int gw = blockIdx.x * 4 + w;
    const int b  = gw / (H_ * 32);
    const int r1 = gw - b * (H_ * 32);
    const int h  = r1 >> 5;
    const int qs0 = (r1 & 31) * 16;

    // Q A-fragments (loop invariant), halves of HD=64
    v16bf qa[2];
#pragma unroll
    for (int half = 0; half < 2; ++half)
        qa[half] = load_frag_g(qkv + (size_t)(b * S_ + qs0 + ln) * TD_
                               + h * HD_ + half * 32 + 8 * sub);

    v8f acc[4] = {};
    float mrow[8], lrow[8];
#pragma unroll
    for (int r = 0; r < 8; ++r) { mrow[r] = -1e30f; lrow[r] = 0.0f; }

    const float* biasbase = ebias + (size_t)h * S_ * S_ + (size_t)(qs0 + sub * 8) * S_;

    for (int kb = 0; kb < S_; kb += 32) {
        // ---- scores: two 16x16 tiles over this 32-key chunk
        v8f sc[2];
#pragma unroll
        for (int t = 0; t < 2; ++t) {
            v8f z = {};
#pragma unroll
            for (int half = 0; half < 2; ++half) {
                v16bf kf = load_frag_g(qkv + (size_t)(b * S_ + kb + t * 16 + ln) * TD_
                                       + D_ + h * HD_ + half * 32 + 8 * sub);
                z = __builtin_amdgcn_wmma_f32_16x16x32_bf16(
                        false, qa[half], false, kf, (short)0, z, false, false);
            }
            sc[t] = z;
        }
        // ---- scale + edge bias
#pragma unroll
        for (int t = 0; t < 2; ++t)
#pragma unroll
            for (int r = 0; r < 8; ++r)
                sc[t][r] = sc[t][r] * 0.125f + biasbase[(size_t)r * S_ + kb + t * 16 + ln];
        // ---- online softmax: DPP16 reductions across the 16 lanes of each half
        float scl[8];
#pragma unroll
        for (int r = 0; r < 8; ++r) {
            const float cm = red16_max(fmaxf(sc[0][r], sc[1][r]));
            const float mn = fmaxf(mrow[r], cm);
            scl[r] = __expf(mrow[r] - mn);
            mrow[r] = mn;
            float p0 = __expf(sc[0][r] - mn);
            float p1 = __expf(sc[1][r] - mn);
            sc[0][r] = p0; sc[1][r] = p1;
            lrow[r] = lrow[r] * scl[r] + red16_sum(p0 + p1);
        }
#pragma unroll
        for (int t = 0; t < 4; ++t)
#pragma unroll
            for (int r = 0; r < 8; ++r)
                acc[t][r] *= scl[r];
        // ---- P: C-layout f32 -> LDS -> A-layout bf16 fragment
#pragma unroll
        for (int t = 0; t < 2; ++t)
#pragma unroll
            for (int r = 0; r < 8; ++r)
                sP[w][(sub * 8 + r) * 32 + t * 16 + ln] = f2bf(sc[t][r]);
        __syncthreads();
        const uint4* pl = reinterpret_cast<const uint4*>(&sP[w][ln * 32 + 8 * sub]);
        Frag32B pfr; pfr.lo = pl[0]; pfr.hi = pl[2];
        v16bf pa = __builtin_bit_cast(v16bf, pfr);
        __syncthreads();
        // ---- O += P * V  (V^T layout: [B,H,HD,S], key dim contiguous)
#pragma unroll
        for (int t = 0; t < 4; ++t) {
            v16bf vf = load_frag_g(vt + (size_t)((b * H_ + h) * HD_ + t * 16 + ln) * S_
                                   + kb + 8 * sub);
            acc[t] = __builtin_amdgcn_wmma_f32_16x16x32_bf16(
                        false, pa, false, vf, (short)0, acc[t], false, false);
        }
    }
    // ---- normalize and emit bf16 O into [B,S,D]
#pragma unroll
    for (int r = 0; r < 8; ++r) {
        const float inv = 1.0f / lrow[r];
        const int s = qs0 + sub * 8 + r;
#pragma unroll
        for (int t = 0; t < 4; ++t)
            o[(size_t)(b * S_ + s) * D_ + h * HD_ + t * 16 + ln] = f2bf(acc[t][r] * inv);
    }
}

// ---------------------------------------------------------------- host
extern "C" void kernel_launch(void* const* d_in, const int* in_sizes, int n_in,
                              void* d_out, int out_size, void* d_ws, size_t ws_size,
                              hipStream_t stream)
{
    (void)in_sizes; (void)n_in; (void)out_size; (void)ws_size;
    const float* x      = (const float*)d_in[0];
    const float* ebias  = (const float*)d_in[1];
    const float* ln1_g  = (const float*)d_in[2];
    const float* ln1_b  = (const float*)d_in[3];
    const float* qkv_w  = (const float*)d_in[4];
    const float* qkv_b  = (const float*)d_in[5];
    const float* proj_w = (const float*)d_in[6];
    const float* proj_b = (const float*)d_in[7];
    const float* ln2_g  = (const float*)d_in[8];
    const float* ln2_b  = (const float*)d_in[9];
    const float* fc1_w  = (const float*)d_in[10];
    const float* fc1_b  = (const float*)d_in[11];
    const float* fc2_w  = (const float*)d_in[12];
    const float* fc2_b  = (const float*)d_in[13];
    float* out = (float*)d_out;

    // workspace layout (bytes)
    char* ws = (char*)d_ws;
    const size_t n_qkvw = (size_t)TD_ * D_;
    const size_t n_projw = (size_t)D_ * D_;
    const size_t n_fc1w = (size_t)FF_ * D_;
    const size_t n_fc2w = (size_t)D_ * FF_;
    const size_t n_act  = (size_t)B_ * S_ * D_;
    const size_t n_qkv  = (size_t)B_ * S_ * TD_;
    const size_t n_vt   = n_act;
    const size_t n_fc1o = (size_t)B_ * S_ * FF_;

    size_t off = 0;
    unsigned short* qkvw_bf = (unsigned short*)(ws + off); off += n_qkvw * 2;
    unsigned short* projw_bf= (unsigned short*)(ws + off); off += n_projw * 2;
    unsigned short* fc1w_bf = (unsigned short*)(ws + off); off += n_fc1w * 2;
    unsigned short* fc2w_bf = (unsigned short*)(ws + off); off += n_fc2w * 2;
    unsigned short* act_bf  = (unsigned short*)(ws + off); off += n_act * 2;   // xn -> o -> h
    unsigned short* qkv_bf  = (unsigned short*)(ws + off); off += n_qkv * 2;
    unsigned short* vt_bf   = (unsigned short*)(ws + off); off += n_vt * 2;
    unsigned short* fc1o_bf = (unsigned short*)(ws + off); off += n_fc1o * 2;

    // 1) weights -> bf16
    cast_f32_bf16<<<(int)((n_qkvw + 255) / 256), 256, 0, stream>>>(qkv_w,  qkvw_bf, (int)n_qkvw);
    cast_f32_bf16<<<(int)((n_projw + 255) / 256), 256, 0, stream>>>(proj_w, projw_bf,(int)n_projw);
    cast_f32_bf16<<<(int)((n_fc1w + 255) / 256), 256, 0, stream>>>(fc1_w,  fc1w_bf, (int)n_fc1w);
    cast_f32_bf16<<<(int)((n_fc2w + 255) / 256), 256, 0, stream>>>(fc2_w,  fc2w_bf, (int)n_fc2w);

    const int tokens = B_ * S_; // 16384
    // 2) LN1
    layernorm_bf16<<<tokens, 256, 0, stream>>>(x, ln1_g, ln1_b, act_bf);
    // 3) QKV GEMM -> bf16 [B,S,3D]
    gemm_bf16_wmma<0><<<dim3(TD_ / 128, tokens / 128), 128, 0, stream>>>(
        act_bf, qkvw_bf, qkv_b, nullptr, nullptr, qkv_bf, tokens, TD_, D_);
    // 4) V transpose
    transpose_v<<<(int)(n_vt / 256), 256, 0, stream>>>(qkv_bf, vt_bf);
    // 5) attention -> bf16 O (reuse act buffer)
    attn_wmma<<<(B_ * H_ * (S_ / 16)) / 4, 128, 0, stream>>>(qkv_bf, vt_bf, ebias, act_bf);
    // 6) proj GEMM + residual(x) -> f32 x2 in d_out
    gemm_bf16_wmma<1><<<dim3(D_ / 128, tokens / 128), 128, 0, stream>>>(
        act_bf, projw_bf, proj_b, x, out, nullptr, tokens, D_, D_);
    // 7) LN2
    layernorm_bf16<<<tokens, 256, 0, stream>>>(out, ln2_g, ln2_b, act_bf);
    // 8) FC1 GEMM + GELU -> bf16
    gemm_bf16_wmma<2><<<dim3(FF_ / 128, tokens / 128), 128, 0, stream>>>(
        act_bf, fc1w_bf, fc1_b, nullptr, nullptr, fc1o_bf, tokens, FF_, D_);
    // 9) FC2 GEMM + residual(x2) -> final f32 in d_out
    gemm_bf16_wmma<1><<<dim3(D_ / 128, tokens / 128), 128, 0, stream>>>(
        fc1o_bf, fc2w_bf, fc2_b, out, out, nullptr, tokens, D_, FF_);
}
